// SUQ_Linear_Kron_62345745269513
// MI455X (gfx1250) — compile-verified, matrix operand order checked by reading, not run on hardware
//
#include <hip/hip_runtime.h>

typedef __attribute__((ext_vector_type(16))) __bf16 v16bf;
typedef __attribute__((ext_vector_type(8)))  __bf16 v8bf;
typedef __attribute__((ext_vector_type(8)))  float  v8f;
typedef __attribute__((ext_vector_type(8)))  unsigned short usv8;
typedef __attribute__((ext_vector_type(4)))  unsigned short usv4;

#define D   1024
#define NB  64
#define TM  128   // workgroup M tile
#define TN  128   // workgroup N tile
#define TK  64    // K staging step (two bf16 WMMA K-steps)
#define NT  (D / TK)
#define LDT 72    // LDS row stride in ushorts (64 + 8 pad, 144 B = 9*16 B)

__device__ __forceinline__ unsigned short f2bf(float f) {
  unsigned int u = __builtin_bit_cast(unsigned int, f);
  unsigned int r = u + 0x7fffu + ((u >> 16) & 1u);   // round-to-nearest-even
  return (unsigned short)(r >> 16);
}

// One lane's 16x32 bf16 fragment from an LDS row (K-contiguous).
// Lanes 0-15 carry K = {0..7, 16..23}; lanes 16-31 carry K = {8..15, 24..31}.
__device__ __forceinline__ v16bf frag_ld(const unsigned short* p, int ks) {
  union { v16bf v; v8bf h[2]; } u;
  u.h[0] = *(const v8bf*)(p + ks);
  u.h[1] = *(const v8bf*)(p + 16 + ks);
  return u.v;
}

__device__ __forceinline__ v8f wmma_bf16(v16bf a, v16bf b, v8f c) {
  return __builtin_amdgcn_wmma_f32_16x16x32_bf16(false, a, false, b, (short)0, c,
                                                 false, false);
}

// ---------------- small kernels ----------------

__global__ __launch_bounds__(256) void k_wcvt(const float* __restrict__ W,
                                              unsigned short* __restrict__ Wbf) {
  size_t gid = (size_t)blockIdx.x * 256 + threadIdx.x;   // D*D/4 threads
  float4 f = *(const float4*)(W + gid * 4);
  usv4 o;
  o[0] = f2bf(f.x); o[1] = f2bf(f.y); o[2] = f2bf(f.z); o[3] = f2bf(f.w);
  *(usv4*)(Wbf + gid * 4) = o;
}

__global__ void k_cinit(float* coef) {
  if (threadIdx.x < NB) coef[threadIdx.x] = 0.f;
}

__global__ __launch_bounds__(256) void k_hmean(const float* __restrict__ a_mean,
                                               const float* __restrict__ W,
                                               const float* __restrict__ bias,
                                               float* __restrict__ out) {
  int gid = blockIdx.x * 256 + threadIdx.x;   // NB*D threads
  int n = gid >> 10, k = gid & (D - 1);
  const float* a = a_mean + (size_t)n * D;
  const float* w = W + (size_t)k * D;
  float s = 0.f;
  for (int j = 0; j < D; j += 4) {
    float4 av = *(const float4*)(a + j);
    float4 wv = *(const float4*)(w + j);
    s += av.x * wv.x + av.y * wv.y + av.z * wv.z + av.w * wv.w;
  }
  out[gid] = s + bias[k];
}

// a_cov chunk f32 -> bf16, fused with coef[n] += sum_ij (am_i*am_j + cov_ij)*A_ij
__global__ __launch_bounds__(256) void k_covcvt(const float* __restrict__ ac,
                                                const float* __restrict__ am,
                                                const float* __restrict__ Akf,
                                                unsigned short* __restrict__ Abf,
                                                float* __restrict__ coef) {
  const int n = blockIdx.y;
  const float* a = ac + (size_t)n * D * D;
  unsigned short* o = Abf + (size_t)n * D * D;
  const float* amn = am + (size_t)n * D;
  const int base = blockIdx.x * 4096;          // 4 rows of 1024 per block
  float s = 0.f;
#pragma unroll
  for (int it = 0; it < 4; ++it) {
    int idx = base + it * 1024 + threadIdx.x * 4;
    float4 cv = *(const float4*)(a + idx);
    float4 av = *(const float4*)(Akf + idx);
    float ai = amn[idx >> 10];
    float4 aj = *(const float4*)(amn + (idx & (D - 1)));
    s += (ai * aj.x + cv.x) * av.x + (ai * aj.y + cv.y) * av.y +
         (ai * aj.z + cv.z) * av.z + (ai * aj.w + cv.w) * av.w;
    usv4 ov;
    ov[0] = f2bf(cv.x); ov[1] = f2bf(cv.y); ov[2] = f2bf(cv.z); ov[3] = f2bf(cv.w);
    *(usv4*)(o + idx) = ov;
  }
  __shared__ float red[256];
  red[threadIdx.x] = s;
  __syncthreads();
  for (int off = 128; off > 0; off >>= 1) {
    if (threadIdx.x < off) red[threadIdx.x] += red[threadIdx.x + off];
    __syncthreads();
  }
  if (threadIdx.x == 0) atomicAdd(&coef[n], red[0]);
}

// ---------------- pipelined GEMM mainloop (shared by both stages) ----------------
// Each wave: 32(M) x 64(N) = 2x4 fragments, 16 WMMAs per 64-wide K step.
// Double-buffered LDS; tile t+1 staged in VGPRs while computing tile t.

#define GEMM_MAINLOOP(ASRC, BSRC)                                              \
  const int tid = threadIdx.x;                                                 \
  const int wid = tid >> 5, lane = tid & 31;                                   \
  const int wm = wid & 3, wn = wid >> 2;                                       \
  const int lh = lane & 15, ks = (lane >> 4) << 3;                             \
  const int lr = tid >> 1, lc = (tid & 1) * 32;                                \
  v8f acc[2][4] = {};                                                          \
  unsigned short* adst = &As[0][lr * LDT + lc];                                \
  unsigned short* bdst = &Bs[0][lr * LDT + lc];                                \
  {                                                                            \
    _Pragma("unroll") for (int e = 0; e < 4; ++e) {                            \
      *(usv8*)(adst + e * 8) = *(const usv8*)(ASRC + e * 8);                   \
      *(usv8*)(bdst + e * 8) = *(const usv8*)(BSRC + e * 8);                   \
    }                                                                          \
  }                                                                            \
  __syncthreads();                                                             \
  for (int t = 0; t < NT; ++t) {                                               \
    const int cur = t & 1;                                                     \
    usv8 an[4], bn[4];                                                         \
    if (t + 1 < NT) {                                                          \
      const unsigned short* as2 = ASRC + (t + 1) * TK;                         \
      const unsigned short* bs2 = BSRC + (t + 1) * TK;                         \
      _Pragma("unroll") for (int e = 0; e < 4; ++e) {                          \
        an[e] = *(const usv8*)(as2 + e * 8);                                   \
        bn[e] = *(const usv8*)(bs2 + e * 8);                                   \
      }                                                                        \
    }                                                                          \
    const unsigned short* Ab = &As[cur][(wm * 32 + lh) * LDT];                 \
    const unsigned short* Bb = &Bs[cur][(wn * 64 + lh) * LDT];                 \
    _Pragma("unroll") for (int kh = 0; kh < TK; kh += 32) {                    \
      v16bf a0 = frag_ld(Ab + kh, ks);                                         \
      v16bf a1 = frag_ld(Ab + 16 * LDT + kh, ks);                              \
      v16bf b0 = frag_ld(Bb + kh, ks);                                         \
      v16bf b1 = frag_ld(Bb + 16 * LDT + kh, ks);                              \
      v16bf b2 = frag_ld(Bb + 32 * LDT + kh, ks);                              \
      v16bf b3 = frag_ld(Bb + 48 * LDT + kh, ks);                              \
      acc[0][0] = wmma_bf16(a0, b0, acc[0][0]);                                \
      acc[1][0] = wmma_bf16(a1, b0, acc[1][0]);                                \
      acc[0][1] = wmma_bf16(a0, b1, acc[0][1]);                                \
      acc[1][1] = wmma_bf16(a1, b1, acc[1][1]);                                \
      acc[0][2] = wmma_bf16(a0, b2, acc[0][2]);                                \
      acc[1][2] = wmma_bf16(a1, b2, acc[1][2]);                                \
      acc[0][3] = wmma_bf16(a0, b3, acc[0][3]);                                \
      acc[1][3] = wmma_bf16(a1, b3, acc[1][3]);                                \
    }                                                                          \
    if (t + 1 < NT) {                                                          \
      unsigned short* ad2 = &As[1 - cur][lr * LDT + lc];                       \
      unsigned short* bd2 = &Bs[1 - cur][lr * LDT + lc];                       \
      _Pragma("unroll") for (int e = 0; e < 4; ++e) {                          \
        *(usv8*)(ad2 + e * 8) = an[e];                                         \
        *(usv8*)(bd2 + e * 8) = bn[e];                                         \
      }                                                                        \
      __syncthreads();                                                         \
    }                                                                          \
  }

// ---------------- stage 1: Tt[n][l][i] = (a_cov_n @ W^T)^T  (bf16) ----------------

__global__ __launch_bounds__(256) void k_gemm1(const unsigned short* __restrict__ Abf,
                                               const unsigned short* __restrict__ Wbf,
                                               unsigned short* __restrict__ Tt) {
  __shared__ __align__(16) unsigned short As[2][TM * LDT];
  __shared__ __align__(16) unsigned short Bs[2][TN * LDT];
  const size_t n = blockIdx.z;
  const unsigned short* Ag = Abf + n * (size_t)(D * D);
  unsigned short* Tg = Tt + n * (size_t)(D * D);
  const int iB = blockIdx.y * TM;   // M = i
  const int lB = blockIdx.x * TN;   // N = l

  const unsigned short* asrc0 =
      Ag + (size_t)(iB + ((threadIdx.x >> 1))) * D + (threadIdx.x & 1) * 32;
  const unsigned short* bsrc0 =
      Wbf + (size_t)(lB + ((threadIdx.x >> 1))) * D + (threadIdx.x & 1) * 32;

  GEMM_MAINLOOP(asrc0, bsrc0)

  // Transposed bf16 store: each lane holds 8 consecutive M(=i) at fixed N(=l).
  const int m0 = (lane >> 4) << 3;
#pragma unroll
  for (int mi = 0; mi < 2; ++mi)
#pragma unroll
    for (int ni = 0; ni < 4; ++ni) {
      int i0 = iB + wm * 32 + mi * 16 + m0;
      int l = lB + wn * 64 + ni * 16 + lh;
      usv8 o;
#pragma unroll
      for (int r = 0; r < 8; ++r) o[r] = f2bf(acc[mi][ni][r]);
      *(usv8*)&Tg[(size_t)l * D + i0] = o;
    }
}

// ---------------- stage 2: h_cov[n] = W @ T_n + coef_n*Bk + b_cov ----------------

__global__ __launch_bounds__(256) void k_gemm2(const unsigned short* __restrict__ Wbf,
                                               const unsigned short* __restrict__ Tt,
                                               const float* __restrict__ coef,
                                               const float* __restrict__ wkB,
                                               const float* __restrict__ bcov,
                                               float* __restrict__ Out) {
  __shared__ __align__(16) unsigned short As[2][TM * LDT];
  __shared__ __align__(16) unsigned short Bs[2][TN * LDT];
  const size_t n = blockIdx.z;
  const unsigned short* Tg = Tt + n * (size_t)(D * D);
  float* Og = Out + n * (size_t)(D * D);
  const int kB = blockIdx.y * TM;   // M = k
  const int lB = blockIdx.x * TN;   // N = l

  const unsigned short* asrc0 =
      Wbf + (size_t)(kB + ((threadIdx.x >> 1))) * D + (threadIdx.x & 1) * 32;
  const unsigned short* bsrc0 =
      Tg + (size_t)(lB + ((threadIdx.x >> 1))) * D + (threadIdx.x & 1) * 32;

  GEMM_MAINLOOP(asrc0, bsrc0)

  const float cf = coef[n];
  const int m0 = (lane >> 4) << 3;
#pragma unroll
  for (int mi = 0; mi < 2; ++mi)
#pragma unroll
    for (int ni = 0; ni < 4; ++ni) {
      int k0 = kB + wm * 32 + mi * 16 + m0;
      int l = lB + wn * 64 + ni * 16 + lh;
#pragma unroll
      for (int r = 0; r < 8; ++r) {
        size_t o = (size_t)(k0 + r) * D + l;
        Og[o] = acc[mi][ni][r] + cf * wkB[o] + bcov[o];
      }
    }
}

// ---------------- launcher ----------------

extern "C" void kernel_launch(void* const* d_in, const int* in_sizes, int n_in,
                              void* d_out, int out_size, void* d_ws, size_t ws_size,
                              hipStream_t stream) {
  const float* a_mean = (const float*)d_in[0];
  const float* a_cov  = (const float*)d_in[1];
  const float* weight = (const float*)d_in[2];
  const float* bias   = (const float*)d_in[3];
  const float* wkA    = (const float*)d_in[4];
  const float* wkB    = (const float*)d_in[5];
  const float* b_cov  = (const float*)d_in[6];

  float* h_mean = (float*)d_out;
  float* h_cov  = h_mean + (size_t)NB * D;

  char* ws = (char*)d_ws;
  float* coef         = (float*)ws;                    // 256 B
  unsigned short* Wbf = (unsigned short*)(ws + 256);   // 2 MB

  // per-chunk workspace: Abf (bf16 a_cov chunk) + Tt (bf16 intermediate), 4 MB/n
  size_t fixed = 256 + (size_t)2 * D * D;
  size_t per_n = (size_t)4 * D * D;
  int chunk = 1;
  if (ws_size > fixed) {
    size_t c = (ws_size - fixed) / per_n;
    if (c > NB) c = NB;
    int p = 1;
    while ((size_t)(p * 2) <= c) p *= 2;
    if (c >= 1) chunk = p;
  }
  unsigned short* Abf = (unsigned short*)(ws + fixed);
  unsigned short* Tt  = Abf + (size_t)chunk * D * D;

  k_wcvt<<<dim3(D * D / (4 * 256)), dim3(256), 0, stream>>>(weight, Wbf);
  k_cinit<<<dim3(1), dim3(64), 0, stream>>>(coef);
  k_hmean<<<dim3(NB * D / 256), dim3(256), 0, stream>>>(a_mean, weight, bias, h_mean);

  for (int n0 = 0; n0 < NB; n0 += chunk) {
    k_covcvt<<<dim3(256, chunk), dim3(256), 0, stream>>>(
        a_cov + (size_t)n0 * D * D, a_mean + (size_t)n0 * D, wkA, Abf, coef + n0);
    dim3 g(D / TN, D / TM, chunk);
    k_gemm1<<<g, dim3(256), 0, stream>>>(Abf, Wbf, Tt);
    k_gemm2<<<g, dim3(256), 0, stream>>>(Wbf, Tt, coef + n0, wkB, b_cov,
                                         h_cov + (size_t)n0 * D * D);
  }
}
